// ModelParallelMaskedMSA_19937238188680
// MI455X (gfx1250) — matrix operand order repaired and checked by reading, not compiled, not run on hardware
//
#include <hip/hip_runtime.h>
#include <hip/hip_bf16.h>

typedef _Float16 f16;
typedef __attribute__((ext_vector_type(16))) _Float16 v16h;
typedef __attribute__((ext_vector_type(8)))  _Float16 v8h;
typedef __attribute__((ext_vector_type(8)))  float    v8f;
typedef __attribute__((ext_vector_type(4)))  unsigned int v4u;
typedef __attribute__((ext_vector_type(8)))  int v8i_t;
typedef __attribute__((ext_vector_type(4)))  int v4i_t;

#define B_   2
#define S_   2048
#define E_   1024
#define H_   16
#define HD_  64
#define E3_  3072

#if defined(__has_builtin)
#if __has_builtin(__builtin_amdgcn_tensor_load_to_lds)
#define HAVE_TDM 1
#endif
#endif

// ---------- helpers -------------------------------------------------------

__device__ inline v16h combine8(v8h lo, v8h hi) {
    v16h r;
#pragma unroll
    for (int i = 0; i < 8; ++i) { r[i] = lo[i]; r[i + 8] = hi[i]; }
    return r;
}

__device__ inline v8f wmma_f16(v16h a, v16h b, v8f c) {
    // 8 args: (neg_a, A, neg_b, B, c_mod, C, reuse_a, reuse_b)
    return __builtin_amdgcn_wmma_f32_16x16x32_f16(false, a, false, b, (short)0, c,
                                                  false, false);
}

__device__ inline float rmax16(float v) {
    v = fmaxf(v, __shfl_xor(v, 1, 32));
    v = fmaxf(v, __shfl_xor(v, 2, 32));
    v = fmaxf(v, __shfl_xor(v, 4, 32));
    v = fmaxf(v, __shfl_xor(v, 8, 32));
    return v;
}
__device__ inline float rsum16(float v) {
    v += __shfl_xor(v, 1, 32);
    v += __shfl_xor(v, 2, 32);
    v += __shfl_xor(v, 4, 32);
    v += __shfl_xor(v, 8, 32);
    return v;
}

// ---------- Kernel 1: QKV = X * W + bias, stored as f16 -------------------
// Double-buffered LDS + register pipelining: one barrier per K-step.

#define TM 128
#define TN 64
#define TK 32

__global__ __launch_bounds__(256) void qkv_gemm_kernel(
        const float* __restrict__ X, const float* __restrict__ W,
        const float* __restrict__ bias, f16* __restrict__ qkvh)
{
    __shared__ __align__(64) _Float16 sA [2][TM][TK];   // 16 KB
    __shared__ __align__(64) _Float16 sBt[2][TN][TK];   //  8 KB, [n][k]

    const int tid  = threadIdx.x;
    const int wave = tid >> 5;
    const int lane = tid & 31;
    const int m0   = blockIdx.y * TM;
    const int n0   = blockIdx.x * TN;

    const int ml     = lane & 15;
    const int half16 = lane >> 4;
    const int kh     = half16 << 3;
    const int kb     = half16 << 4;

    v8f acc[4];
#pragma unroll
    for (int t = 0; t < 4; ++t)
#pragma unroll
        for (int i = 0; i < 8; ++i) acc[t][i] = 0.0f;

    float4 ra[4], rw[2];
    auto load_stage = [&](int k0) {
#pragma unroll
        for (int c = 0; c < 4; ++c) {
            int idx = tid + c * 256;
            int r   = idx >> 3;
            int cc  = (idx & 7) * 4;
            ra[c] = *(const float4*)(X + (size_t)(m0 + r) * E_ + k0 + cc);
        }
#pragma unroll
        for (int c = 0; c < 2; ++c) {
            int idx = tid + c * 256;
            int kk  = idx >> 4;
            int nn  = (idx & 15) * 4;
            rw[c] = *(const float4*)(W + (size_t)(k0 + kk) * E3_ + n0 + nn);
        }
    };
    auto store_stage = [&](int bf) {
#pragma unroll
        for (int c = 0; c < 4; ++c) {
            int idx = tid + c * 256;
            int r   = idx >> 3;
            int cc  = (idx & 7) * 4;
            sA[bf][r][cc + 0] = (_Float16)ra[c].x;
            sA[bf][r][cc + 1] = (_Float16)ra[c].y;
            sA[bf][r][cc + 2] = (_Float16)ra[c].z;
            sA[bf][r][cc + 3] = (_Float16)ra[c].w;
        }
#pragma unroll
        for (int c = 0; c < 2; ++c) {
            int idx = tid + c * 256;
            int kk  = idx >> 4;
            int nn  = (idx & 15) * 4;
            sBt[bf][nn + 0][kk] = (_Float16)rw[c].x;
            sBt[bf][nn + 1][kk] = (_Float16)rw[c].y;
            sBt[bf][nn + 2][kk] = (_Float16)rw[c].z;
            sBt[bf][nn + 3][kk] = (_Float16)rw[c].w;
        }
    };

    load_stage(0);
    int buf = 0;
    for (int k0 = 0; k0 < E_; k0 += TK, buf ^= 1) {
        store_stage(buf);
        if (k0 + TK < E_) load_stage(k0 + TK);
        __syncthreads();
        v8h lo = *(const v8h*)(&sA[buf][wave * 16 + ml][kh]);
        v8h hi = *(const v8h*)(&sA[buf][wave * 16 + ml][16 + kh]);
        v16h afrag = combine8(lo, hi);
#pragma unroll
        for (int t = 0; t < 4; ++t) {
            v16h bfrag = combine8(*(const v8h*)(&sBt[buf][t * 16 + ml][kh]),
                                  *(const v8h*)(&sBt[buf][t * 16 + ml][16 + kh]));
            acc[t] = wmma_f16(afrag, bfrag, acc[t]);
        }
    }

    const int mbase = m0 + wave * 16 + (half16 << 3);
#pragma unroll
    for (int t = 0; t < 4; ++t) {
        int col  = n0 + t * 16 + ml;
        float bv = bias[col];
#pragma unroll
        for (int i = 0; i < 8; ++i) {
            int row = mbase + i;
            qkvh[(size_t)row * E3_ + col] = (_Float16)(acc[t][i] + bv);
        }
    }
}

// ---------- Kernel 2: causal flash attention ------------------------------
// The reference's reshape is LINEAR in t:  Q[t,c] = qkvh[b][h*393216 + t*192 + c]
// (K: +64, V: +128), and out[t,c] = out_flat[b*S*E + h*131072 + t*64 + c].
// grid = (16 query tiles of 128, B*H), block = 256 (8 waves x 16 query rows).
// K tile (32x64, row stride 192) staged by ONE tensor_load_to_lds per block
// (TENSORcnt); V tile staged per-lane with global_load_async_to_lds_b128
// (ASYNCcnt). Double-buffered, one barrier per 32-key block.

__global__ __launch_bounds__(256) void flash_attn_kernel(
        const f16* __restrict__ qkvh, float* __restrict__ out)
{
    __shared__ __align__(64) _Float16 sK [2][32][HD_];   // [key][c]   8 KB
    __shared__ __align__(64) _Float16 sVr[2][32][HD_];   // [key][c]   8 KB
    __shared__ __align__(64) _Float16 sP [8][16][32];    // per-wave P 8 KB

    const int tid  = threadIdx.x;
    const int wave = tid >> 5;
    const int lane = tid & 31;
    const int bh   = blockIdx.y;
    const int b    = bh >> 4;
    const int h    = bh & 15;
    const int qt   = blockIdx.x;
    const int tw   = qt * 128 + wave * 16;      // wave's first query row

    const f16* hbase  = qkvh + (size_t)b * (S_ * E3_) + (size_t)h * (128 * E3_);
    const float scale = 0.022097086912079608f;  // 1/sqrt(2*1024)

    const int ml     = lane & 15;
    const int half16 = lane >> 4;
    const int kh     = half16 << 3;
    const int kb     = half16 << 4;

    // per-thread V staging coords: one 16B chunk per block
    const int skk = tid >> 3;          // key row in tile, 0..31
    const int sc0 = (tid & 7) * 8;     // channel start, 0..56

    auto issue_stage = [&](int kbk, int bf) {
        // V tile: per-lane async copy to LDS (ASYNCcnt)
        {
            unsigned long long ga = (unsigned long long)(uintptr_t)
                (hbase + (size_t)(kbk * 32 + skk) * 192 + 128 + sc0);
            unsigned lds = (unsigned)(uintptr_t)(&sVr[bf][skk][sc0]);
            asm volatile("global_load_async_to_lds_b128 %0, %1, off"
                         :: "v"(lds), "v"(ga) : "memory");
        }
#ifdef HAVE_TDM
        // K tile: one TDM descriptor: 32 rows x 64 halves, row stride 192
        if (wave == 0) {
            unsigned long long ga = (unsigned long long)(uintptr_t)
                (hbase + (size_t)(kbk * 32) * 192 + 64);
            unsigned lds = (unsigned)(uintptr_t)(&sK[bf][0][0]);
            v4u g0 = { 1u,                                    // count=1
                       lds,                                   // lds_addr
                       (unsigned)ga,                          // global_addr lo
                       (unsigned)((ga >> 32) & 0x01FFFFFFull) // global_addr hi
                           | (2u << 30) };                    // type=2 (image)
            v8i_t g1 = { 0x00010000,          // data_size = 2 bytes
                         (int)(64u << 16),    // tensor_dim0 = 64
                         (int)(0xFFFFu << 16),// tensor_dim1 = 0xFFFF
                         (int)(64u << 16),    // tile_dim0 = 64
                         32,                  // tile_dim1 = 32 (tile_dim2 = 0)
                         192,                 // tensor_dim0_stride = 192
                         0, 0 };
            v4i_t gz  = { 0, 0, 0, 0 };
            v8i_t gz8 = { 0, 0, 0, 0, 0, 0, 0, 0 };
            __builtin_amdgcn_tensor_load_to_lds(g0, g1, gz, gz, gz8, 0);
        }
#else
        // fallback: per-lane async copy of K tile
        {
            unsigned long long ga = (unsigned long long)(uintptr_t)
                (hbase + (size_t)(kbk * 32 + skk) * 192 + 64 + sc0);
            unsigned lds = (unsigned)(uintptr_t)(&sK[bf][skk][sc0]);
            asm volatile("global_load_async_to_lds_b128 %0, %1, off"
                         :: "v"(lds), "v"(ga) : "memory");
        }
#endif
    };

    // ---- Q fragments (persist across key blocks) ----
    const f16* qrow = hbase + (size_t)(tw + ml) * 192;
    v16h q0 = *(const v16h*)(qrow + kb);
    v16h q1 = *(const v16h*)(qrow + 32 + kb);

    float mrun[8], lrun[8];
    v8f acc[4];
#pragma unroll
    for (int i = 0; i < 8; ++i) { mrun[i] = -1e30f; lrun[i] = 0.0f; }
#pragma unroll
    for (int t = 0; t < 4; ++t)
#pragma unroll
        for (int i = 0; i < 8; ++i) acc[t][i] = 0.0f;

    v8f zc;
#pragma unroll
    for (int i = 0; i < 8; ++i) zc[i] = 0.0f;

    issue_stage(0, 0);

    const int nblocks = (qt + 1) * 4;           // causal: keys up to qt*128+127
    for (int kbk = 0; kbk < nblocks; ++kbk) {
        const int buf   = kbk & 1;
        const int kbase = kbk * 32;

        // this block's staging must have landed in LDS
        asm volatile("s_wait_asynccnt 0" ::: "memory");
#ifdef HAVE_TDM
        if (wave == 0) __builtin_amdgcn_s_wait_tensorcnt(0);
#endif
        __syncthreads();

        // start staging the NEXT block while we compute this one
        if (kbk + 1 < nblocks) issue_stage(kbk + 1, buf ^ 1);

        // ---- S = Q * K^T : keys 0..15 (sa) and 16..31 (sb) ----
        v16h kA0 = *(const v16h*)(&sK[buf][ml][kb]);
        v16h kA1 = *(const v16h*)(&sK[buf][ml][32 + kb]);
        v16h kB0 = *(const v16h*)(&sK[buf][16 + ml][kb]);
        v16h kB1 = *(const v16h*)(&sK[buf][16 + ml][32 + kb]);
        v8f sa = wmma_f16(q1, kA1, wmma_f16(q0, kA0, zc));
        v8f sb = wmma_f16(q1, kB1, wmma_f16(q0, kB0, zc));

        // ---- online softmax (row m = i + 8*half16) ----
        const int keyA = kbase + ml;
        const int keyB = keyA + 16;
        float corr[8];
#pragma unroll
        for (int i = 0; i < 8; ++i) {
            int trow = tw + i + (half16 << 3);
            bool maskA = keyA > trow, maskB = keyB > trow;
            float va = maskA ? -1e30f : sa[i] * scale;
            float vb = maskB ? -1e30f : sb[i] * scale;
            float rm = rmax16(fmaxf(va, vb));
            float nm = fmaxf(mrun[i], rm);
            corr[i]  = __expf(mrun[i] - nm);
            mrun[i]  = nm;
            float pa = maskA ? 0.0f : __expf(va - nm);
            float pb = maskB ? 0.0f : __expf(vb - nm);
            lrun[i]  = lrun[i] * corr[i] + rsum16(pa + pb);
            int m = i + (half16 << 3);
            sP[wave][m][ml]      = (_Float16)pa;
            sP[wave][m][16 + ml] = (_Float16)pb;
        }
#pragma unroll
        for (int t = 0; t < 4; ++t)
#pragma unroll
            for (int i = 0; i < 8; ++i) acc[t][i] *= corr[i];

        // P was written cross-lane via LDS; wait before reading back
        asm volatile("s_wait_dscnt 0" ::: "memory");

        // ---- O += P * V ----  (V B-frags gathered strided from row-major sVr)
        v16h pf = *(const v16h*)(&sP[wave][ml][kb]);
#pragma unroll
        for (int t = 0; t < 4; ++t) {
            v16h vf;
#pragma unroll
            for (int j = 0; j < 16; ++j) vf[j] = sVr[buf][kb + j][t * 16 + ml];
            acc[t] = wmma_f16(pf, vf, acc[t]);
        }
    }

    // ---- epilogue: normalize, linear output mapping out += t*64 + c ----
    const size_t obase = (size_t)b * (S_ * E_) + (size_t)h * (128 * E_);
#pragma unroll
    for (int t = 0; t < 4; ++t) {
#pragma unroll
        for (int i = 0; i < 8; ++i) {
            int m = i + (half16 << 3);
            out[obase + (size_t)(tw + m) * HD_ + t * 16 + ml] = acc[t][i] / lrun[i];
        }
    }
}

// ---------- launch --------------------------------------------------------

extern "C" void kernel_launch(void* const* d_in, const int* in_sizes, int n_in,
                              void* d_out, int out_size, void* d_ws, size_t ws_size,
                              hipStream_t stream) {
    const float* X    = (const float*)d_in[0];   // [B,S,E]   f32
    const float* W    = (const float*)d_in[1];   // [E,3E]    f32
    const float* bias = (const float*)d_in[2];   // [1,3E]    f32
    float* out        = (float*)d_out;           // [B,S,E]   f32
    f16* qkvh         = (f16*)d_ws;              // [B*S,3E]  f16 (25.2 MB)

    dim3 g1(E3_ / TN, (B_ * S_) / TM);           // 48 x 32
    qkv_gemm_kernel<<<g1, 256, 0, stream>>>(X, W, bias, qkvh);

    dim3 g2(S_ / 128, B_ * H_);                  // 16 x 32
    flash_attn_kernel<<<g2, 256, 0, stream>>>(qkvh, out);
}
